// CrossAttention2D_83021717831729
// MI455X (gfx1250) — compile-verified, matrix operand order checked by reading, not run on hardware
//
#include <hip/hip_runtime.h>
#include <hip/hip_bf16.h>
#include <cstdint>

// CDNA5 / gfx1250 cross-attention: all GEMMs via v_wmma_f32_16x16x32_bf16,
// bf16-copy staging via GLOBAL_LOAD_ASYNC_TO_LDS_B128 (ASYNCcnt) when available.

typedef __bf16 bf16_t;
typedef __attribute__((ext_vector_type(16))) __bf16 v16bf;
typedef __attribute__((ext_vector_type(8)))  float  v8f;

#define DEV_INLINE __device__ __forceinline__

#if __has_builtin(__builtin_amdgcn_global_load_async_to_lds_b128)
#define HAVE_ASYNC_LDS 1
// Exact pointee type from the builtin's prototype (clang diagnostic):
//   __attribute__((__vector_size__(4 * sizeof(int)))) int  in AS1 / AS3
typedef int v4i_vs __attribute__((vector_size(16)));
typedef __attribute__((address_space(1))) v4i_vs gv4i;
typedef __attribute__((address_space(3))) v4i_vs lv4i;
#endif

// Copy 16 contiguous bytes global -> LDS. Async (no VGPR round-trip) on the
// gfx1250 async path; falls back to a regular b128 load/ds_store pair.
DEV_INLINE void copy16(const void* g, void* l) {
#ifdef HAVE_ASYNC_LDS
  __builtin_amdgcn_global_load_async_to_lds_b128(
      (gv4i*)(uintptr_t)g, (lv4i*)(uintptr_t)l, 0, 0);
#else
  *(uint4*)l = *(const uint4*)g;
#endif
}

DEV_INLINE void wait_async() {
#ifdef HAVE_ASYNC_LDS
#if __has_builtin(__builtin_amdgcn_s_wait_asynccnt)
  __builtin_amdgcn_s_wait_asynccnt(0);
#else
  asm volatile("s_wait_asynccnt 0" ::: "memory");
#endif
#endif
}

DEV_INLINE bf16_t f2bf(float f) {
  unsigned u = __builtin_bit_cast(unsigned, f);
  unsigned r = u + 0x7FFFu + ((u >> 16) & 1u);   // round-to-nearest-even
  unsigned short h = (unsigned short)(r >> 16);
  return __builtin_bit_cast(bf16_t, h);
}

// Build a 16x32 bf16 WMMA operand from an LDS tile stored row-major
// (ISA 7.12.2 16-bit A layout; B operand uses same layout on a transposed tile)
DEV_INLINE v16bf load_frag(const bf16_t* __restrict__ t, int row0, int lane,
                           int stride, int kOff) {
  const int l15 = lane & 15;
  const int kb  = (lane >> 4) * 8;
  const bf16_t* p = t + (row0 + l15) * stride + kOff + kb;
  v16bf a;
#pragma unroll
  for (int i = 0; i < 8; ++i) a[i] = p[i];
#pragma unroll
  for (int i = 0; i < 8; ++i) a[8 + i] = p[16 + i];
  return a;
}

DEV_INLINE v8f wmma_bf16(v16bf a, v16bf b, v8f c) {
  return __builtin_amdgcn_wmma_f32_16x16x32_bf16(false, a, false, b,
                                                 (short)0, c, false, false);
}

// ---------------------------------------------------------------------------
// Projection GEMM: D[b,m,n](bf16) = A[b,m,:] @ W[:,n] + bias[n]
// A_T: A stored [B,K,M] (x tensor); else [B,M,K] (context).
// (f32 inputs need conversion, so staging stays on the VGPR path here.)
// ---------------------------------------------------------------------------
template <bool A_T>
__global__ void __launch_bounds__(128)
proj_gemm(const float* __restrict__ A, const float* __restrict__ W,
          const float* __restrict__ bias, bf16_t* __restrict__ D,
          int M, int N, int K) {
  __shared__ bf16_t As[64 * 40];   // [m][k], k-chunk 32, padded stride 40
  __shared__ bf16_t Bs[64 * 40];   // [n][k] (W tile transposed)
  const int b = blockIdx.z;
  const int mBase = blockIdx.y * 64;
  const int nBase = blockIdx.x * 64;
  const int tid = threadIdx.x;
  const int lane = tid & 31, wave = tid >> 5;

  v8f acc[4];
#pragma unroll
  for (int f = 0; f < 4; ++f)
#pragma unroll
    for (int v = 0; v < 8; ++v) acc[f][v] = 0.0f;

  const float* Ab = A + (size_t)b * M * K;
  for (int k0 = 0; k0 < K; k0 += 32) {
#pragma unroll 4
    for (int i = tid; i < 64 * 32; i += 128) {
      int m, k;
      if (A_T) { m = i & 63; k = i >> 6; }   // coalesce over m (contig in A_T)
      else     { k = i & 31; m = i >> 5; }   // coalesce over k
      float v = A_T ? Ab[(size_t)(k0 + k) * M + (mBase + m)]
                    : Ab[(size_t)(mBase + m) * K + (k0 + k)];
      As[m * 40 + k] = f2bf(v);
    }
#pragma unroll 4
    for (int i = tid; i < 64 * 32; i += 128) {
      int k = i >> 6, n = i & 63;            // coalesce over n
      Bs[n * 40 + k] = f2bf(W[(size_t)(k0 + k) * N + (nBase + n)]);
    }
    __syncthreads();
    const v16bf a = load_frag(As, 16 * wave, lane, 40, 0);
#pragma unroll
    for (int f = 0; f < 4; ++f) {
      v16bf bb = load_frag(Bs, 16 * f, lane, 40, 0);
      acc[f] = wmma_bf16(a, bb, acc[f]);
    }
    __syncthreads();
  }

  bf16_t* Db = D + (size_t)b * M * N;
  const int l15 = lane & 15, lhi = lane >> 4;
#pragma unroll
  for (int f = 0; f < 4; ++f) {
    int n = nBase + 16 * f + l15;
    float bvv = bias[n];
#pragma unroll
    for (int v = 0; v < 8; ++v) {
      int m = mBase + 16 * wave + 8 * lhi + v;   // C-layout: VGPRv -> M=v(+8)
      Db[(size_t)m * N + n] = f2bf(acc[f][v] + bvv);
    }
  }
}

// ---------------------------------------------------------------------------
// Attention: per (b, head, 64-query tile) compute softmax(QK^T/sqrt(d)+mask)V.
// Q/K tiles staged with async global->LDS b128 copies (bf16 passthrough).
// ---------------------------------------------------------------------------
__global__ void __launch_bounds__(128)
attn_kernel(const bf16_t* __restrict__ Qb, const bf16_t* __restrict__ Kb,
            const bf16_t* __restrict__ Vb, const int* __restrict__ cmask,
            bf16_t* __restrict__ Ob) {
  constexpr int NTOK = 1024, T = 256, C = 768, HD = 64;
  __shared__ char smem[53248];
  bf16_t* Qs  = (bf16_t*)(smem);            // 64 x 72  [0, 9216)
  bf16_t* Ks  = (bf16_t*)(smem + 16384);    // 128 x 72 [16384, 34816)
  bf16_t* Ps  = (bf16_t*)(smem);            // 64 x 264 [0, 33792) (reuses Q/K)
  bf16_t* Vst = (bf16_t*)(smem + 34816);    // 64 x 136 [34816, 52224) (V^T)
  float*  msk = (float*)(smem + 52224);     // 256 floats

  const int b = blockIdx.z, h = blockIdx.y, mBase = blockIdx.x * 64;
  const int tid = threadIdx.x, lane = tid & 31, wave = tid >> 5;
  const int l15 = lane & 15, lhi = lane >> 4;

  // async-stage Q tile: 64 rows x 8 chunks of 16B (stride 72*2=144B, 16B-mult)
  {
    const bf16_t* Qg = Qb + ((size_t)b * NTOK + mBase) * C + h * HD;
#pragma unroll
    for (int c = tid; c < 512; c += 128) {
      int m = c >> 3, dc = (c & 7) * 8;
      copy16(Qg + (size_t)m * C + dc, Qs + m * 72 + dc);
    }
  }
  for (int t = tid; t < T; t += 128)
    msk[t] = (cmask[b * T + t] == 0) ? -1e30f : 0.0f;

  // ---- scores S[64 x 256]: 16 fragments per wave, K-dim = 64 (2 wmma steps)
  v8f acc[16];
#pragma unroll
  for (int f = 0; f < 16; ++f)
#pragma unroll
    for (int v = 0; v < 8; ++v) acc[f][v] = 0.0f;

  for (int c = 0; c < 2; ++c) {
    __syncthreads();                       // prior chunk's frag reads done
    const bf16_t* Kg = Kb + ((size_t)b * T + 128 * c) * C + h * HD;
#pragma unroll
    for (int i = tid; i < 1024; i += 128) {
      int tt = i >> 3, dc = (i & 7) * 8;
      copy16(Kg + (size_t)tt * C + dc, Ks + tt * 72 + dc);
    }
    wait_async();
    __syncthreads();
#pragma unroll
    for (int kk = 0; kk < 2; ++kk) {
      v16bf a = load_frag(Qs, 16 * wave, lane, 72, 32 * kk);
#pragma unroll
      for (int fl = 0; fl < 8; ++fl) {
        v16bf bb = load_frag(Ks, 16 * fl, lane, 72, 32 * kk);  // rows of K
        acc[8 * c + fl] = wmma_bf16(a, bb, acc[8 * c + fl]);
      }
    }
  }
  __syncthreads();

  // ---- softmax over t: lane holds S[m = 16w+8*lhi+v][t = 16f+l15]
  const float scale = 0.125f;  // 1/sqrt(64)
  float rmax[8], rsum[8];
#pragma unroll
  for (int v = 0; v < 8; ++v) rmax[v] = -3.0e38f;
#pragma unroll
  for (int f = 0; f < 16; ++f) {
    float madd = msk[16 * f + l15];
#pragma unroll
    for (int v = 0; v < 8; ++v) {
      float s = acc[f][v] * scale + madd;
      acc[f][v] = s;
      rmax[v] = fmaxf(rmax[v], s);
    }
  }
#pragma unroll
  for (int sh = 1; sh <= 8; sh <<= 1)            // reduce within 16-lane halves
#pragma unroll
    for (int v = 0; v < 8; ++v)
      rmax[v] = fmaxf(rmax[v], __shfl_xor(rmax[v], sh, 32));
#pragma unroll
  for (int v = 0; v < 8; ++v) rsum[v] = 0.0f;
#pragma unroll
  for (int f = 0; f < 16; ++f)
#pragma unroll
    for (int v = 0; v < 8; ++v) {
      float e = __expf(acc[f][v] - rmax[v]);
      acc[f][v] = e;
      rsum[v] += e;
    }
#pragma unroll
  for (int sh = 1; sh <= 8; sh <<= 1)
#pragma unroll
    for (int v = 0; v < 8; ++v) rsum[v] += __shfl_xor(rsum[v], sh, 32);

#pragma unroll
  for (int v = 0; v < 8; ++v) {
    float inv = 1.0f / rsum[v];
    int m = 16 * wave + 8 * lhi + v;
#pragma unroll
    for (int f = 0; f < 16; ++f)
      Ps[m * 264 + 16 * f + l15] = f2bf(acc[f][v] * inv);
  }
  __syncthreads();

  // ---- O = P @ V : K-dim = 256 (V staged transposed -> VGPR path)
  v8f o[4];
#pragma unroll
  for (int f = 0; f < 4; ++f)
#pragma unroll
    for (int v = 0; v < 8; ++v) o[f][v] = 0.0f;

  for (int c = 0; c < 2; ++c) {
    for (int i = tid; i < 128 * 64; i += 128) {
      int tt = i >> 6, d = i & 63;
      Vst[d * 136 + tt] = Vb[((size_t)b * T + 128 * c + tt) * C + h * HD + d];
    }
    __syncthreads();
#pragma unroll
    for (int kk = 0; kk < 4; ++kk) {
      v16bf a = load_frag(Ps, 16 * wave, lane, 264, 128 * c + 32 * kk);
#pragma unroll
      for (int f = 0; f < 4; ++f) {
        v16bf bb = load_frag(Vst, 16 * f, lane, 136, 32 * kk);
        o[f] = wmma_bf16(a, bb, o[f]);
      }
    }
    __syncthreads();
  }

#pragma unroll
  for (int f = 0; f < 4; ++f) {
    int d = 16 * f + l15;
#pragma unroll
    for (int v = 0; v < 8; ++v) {
      int m = mBase + 16 * wave + 8 * lhi + v;
      Ob[((size_t)b * NTOK + m) * C + h * HD + d] = f2bf(o[f][v]);
    }
  }
}

// ---------------------------------------------------------------------------
// Output projection: Out[b,c,n](f32) = (attn[b,n,:] @ Wo[:,c]) + bo[c]
// A tile (bf16) staged via async b128 copies; stores transposed as b128.
// ---------------------------------------------------------------------------
__global__ void __launch_bounds__(128)
out_gemm(const bf16_t* __restrict__ A, const float* __restrict__ W,
         const float* __restrict__ bias, float* __restrict__ Out,
         int M, int N, int K) {
  __shared__ bf16_t As[64 * 40];
  __shared__ bf16_t Bs[64 * 40];
  const int b = blockIdx.z;
  const int mBase = blockIdx.y * 64, nBase = blockIdx.x * 64;
  const int tid = threadIdx.x, lane = tid & 31, wave = tid >> 5;

  v8f acc[4];
#pragma unroll
  for (int f = 0; f < 4; ++f)
#pragma unroll
    for (int v = 0; v < 8; ++v) acc[f][v] = 0.0f;

  const bf16_t* Ab = A + (size_t)b * M * K;
  for (int k0 = 0; k0 < K; k0 += 32) {
    // 64 rows x 4 chunks of 16B (row pitch 40*2=80B, 16B-mult)
#pragma unroll
    for (int c = tid; c < 256; c += 128) {
      int m = c >> 2, kc = (c & 3) * 8;
      copy16(Ab + (size_t)(mBase + m) * K + k0 + kc, As + m * 40 + kc);
    }
#pragma unroll 4
    for (int i = tid; i < 64 * 32; i += 128) {
      int k = i >> 6, n = i & 63;
      Bs[n * 40 + k] = f2bf(W[(size_t)(k0 + k) * N + (nBase + n)]);
    }
    wait_async();
    __syncthreads();
    const v16bf a = load_frag(As, 16 * wave, lane, 40, 0);
#pragma unroll
    for (int f = 0; f < 4; ++f) {
      v16bf bb = load_frag(Bs, 16 * f, lane, 40, 0);
      acc[f] = wmma_bf16(a, bb, acc[f]);
    }
    __syncthreads();
  }

  float* Ob = Out + (size_t)b * N * M;   // [c][n_token]
  const int l15 = lane & 15, lhi = lane >> 4;
#pragma unroll
  for (int f = 0; f < 4; ++f) {
    int n = nBase + 16 * f + l15;
    float bvv = bias[n];
    int m0 = mBase + 16 * wave + 8 * lhi;
    float4 lo = make_float4(acc[f][0] + bvv, acc[f][1] + bvv,
                            acc[f][2] + bvv, acc[f][3] + bvv);
    float4 hi = make_float4(acc[f][4] + bvv, acc[f][5] + bvv,
                            acc[f][6] + bvv, acc[f][7] + bvv);
    float4* pp = (float4*)(Ob + (size_t)n * M + m0);
    pp[0] = lo;
    pp[1] = hi;
  }
}

// ---------------------------------------------------------------------------
extern "C" void kernel_launch(void* const* d_in, const int* in_sizes, int n_in,
                              void* d_out, int out_size, void* d_ws,
                              size_t ws_size, hipStream_t stream) {
  const float* x   = (const float*)d_in[0];
  const float* ctx = (const float*)d_in[1];
  const int*   cm  = (const int*)d_in[2];
  const float* Wq  = (const float*)d_in[3];
  const float* bq  = (const float*)d_in[4];
  const float* Wk  = (const float*)d_in[5];
  const float* bk  = (const float*)d_in[6];
  const float* Wv  = (const float*)d_in[7];
  const float* bvp = (const float*)d_in[8];
  const float* Wo  = (const float*)d_in[9];
  const float* bo  = (const float*)d_in[10];
  float* out = (float*)d_out;

  constexpr int B = 8, C = 768, NTOK = 1024, T = 256, CTXD = 1024, NH = 12;

  char* ws = (char*)d_ws;
  const size_t qBytes  = (size_t)B * NTOK * C * sizeof(bf16_t);  // 12.6 MB
  const size_t kvBytes = (size_t)B * T * C * sizeof(bf16_t);     // 3.15 MB
  bf16_t* Qb = (bf16_t*)(ws);
  bf16_t* Kb = (bf16_t*)(ws + qBytes);
  bf16_t* Vb = (bf16_t*)(ws + qBytes + kvBytes);
  bf16_t* Ob = (bf16_t*)(ws + qBytes + 2 * kvBytes);

  dim3 blk(128);
  // Q = x^T @ Wq + bq        (x is [B, C, N] -> A transposed)
  proj_gemm<true><<<dim3(C / 64, NTOK / 64, B), blk, 0, stream>>>(
      x, Wq, bq, Qb, NTOK, C, C);
  // K = ctx @ Wk + bk ; V = ctx @ Wv + bv
  proj_gemm<false><<<dim3(C / 64, T / 64, B), blk, 0, stream>>>(
      ctx, Wk, bk, Kb, T, C, CTXD);
  proj_gemm<false><<<dim3(C / 64, T / 64, B), blk, 0, stream>>>(
      ctx, Wv, bvp, Vb, T, C, CTXD);
  // attention per (query-tile, head, batch)
  attn_kernel<<<dim3(NTOK / 64, NH, B), blk, 0, stream>>>(Qb, Kb, Vb, cm, Ob);
  // Out = attn @ Wo + bo, stored transposed to [B, C, H, W]
  out_gemm<<<dim3(C / 64, NTOK / 64, B), blk, 0, stream>>>(
      Ob, Wo, bo, out, NTOK, C, C);
}
